// gnn_31241592111180
// MI455X (gfx1250) — matrix operand order repaired and checked by reading, not compiled, not run on hardware
//
#include <hip/hip_runtime.h>
#include <hip/hip_bf16.h>
#include <math.h>

// ---------------------------------------------------------------------------
// 2-layer GCN for MI455X (gfx1250, wave32).
//   GEMMs: v_wmma_f32_16x16x32_bf16, one 16x16 tile per wave, K=128 -> 4 WMMA.
//   Aggregation: relaxed agent-scope float atomics (L2-resident, 192MB L2).
// ---------------------------------------------------------------------------

typedef __attribute__((ext_vector_type(16))) __bf16 v16bf;
typedef __attribute__((ext_vector_type(8)))  float  v8f;

__device__ __forceinline__ void atomAddF(float* p, float v) {
  __hip_atomic_fetch_add(p, v, __ATOMIC_RELAXED, __HIP_MEMORY_SCOPE_AGENT);
}

// ---- degree / norm prep ----------------------------------------------------
__global__ void k_fill1(float* d, int n) {
  int i = blockIdx.x * blockDim.x + threadIdx.x;
  if (i < n) d[i] = 1.0f;               // self-loop contributes 1 to degree
}

__global__ void k_deg_edge(float* deg, const int* dst, int e) {
  int i = blockIdx.x * blockDim.x + threadIdx.x;
  if (i < e) atomAddF(&deg[dst[i]], 1.0f);
}

__global__ void k_rsqrt(float* d, int n) {
  int i = blockIdx.x * blockDim.x + threadIdx.x;
  if (i < n) d[i] = rsqrtf(d[i]);       // deg >= 1 always (self loops)
}

// ---- weight convert: W[K][Nin] f32 -> WT[Npad][K] bf16 (transposed, padded)
__global__ void k_wt(const float* __restrict__ W, __bf16* __restrict__ WT,
                     int K, int Nin, int Npad) {
  int i = blockIdx.x * blockDim.x + threadIdx.x;
  if (i >= Npad * K) return;
  int n = i / K, k = i % K;
  WT[i] = (n < Nin) ? (__bf16)W[k * Nin + n] : (__bf16)0.0f;
}

// ---- WMMA GEMM: Out[M][LDOUT] = A[M][128] (f32) x WT^T (bf16) --------------
// One wave per 16x16 output tile; blockDim.x/32 waves cover blockDim.x/2 cols.
// LDOUT is compile-time so the epilogue stores use immediate offsets.
template <int LDOUT>
__global__ void gemm_wmma(const float* __restrict__ A,
                          const __bf16* __restrict__ WT,  // [Npad][128]
                          float* __restrict__ Out, int M) {
  __shared__ __bf16 Abf[16 * 128];                 // 4 KB bf16 row strip
  const int tid  = threadIdx.x;
  const int row0 = blockIdx.x * 16;

  // stage 16x128 f32 strip -> bf16 LDS
  for (int i = tid; i < 16 * 128; i += blockDim.x) {
    int r = i >> 7, k = i & 127;
    int rr = row0 + r; if (rr >= M) rr = M - 1;    // clamp for ragged tail
    Abf[i] = (__bf16)A[(size_t)rr * 128 + k];
  }
  __syncthreads();

  const int wave = tid >> 5, lane = tid & 31;
  const int colbase = wave * 16;
  const int lrow = lane & 15, lhi = lane >> 4;     // wave32 lane split
  const __bf16* wrow = WT + (size_t)(colbase + lrow) * 128;

  v8f c = {};
#pragma unroll
  for (int kk = 0; kk < 4; ++kk) {                 // K = 128 = 4 x 32
    const int k0 = kk * 32;
    v16bf a, b;
#pragma unroll
    for (int j = 0; j < 8; ++j) {
      // A 16x32 bf16 layout: lanes 0-15 K in {0..7,16..23}, lanes 16-31 +8
      a[j]     = Abf[lrow * 128 + k0 + lhi * 8 + j];
      a[8 + j] = Abf[lrow * 128 + k0 + 16 + lhi * 8 + j];
      // B 32x16 bf16 layout: lane holds column lrow, K same as A
      b[j]     = wrow[k0 + lhi * 8 + j];
      b[8 + j] = wrow[k0 + 16 + lhi * 8 + j];
    }
    c = __builtin_amdgcn_wmma_f32_16x16x32_bf16(
        /*neg_a=*/false, a, /*neg_b=*/false, b,
        /*c_mod=*/(short)0, c, /*reuse_a=*/false, /*reuse_b=*/false);
  }

  // C/D layout: VGPR v -> row 8*lhi + v, col = lrow.
  // Base pointer for this lane's column; rows at compile-time stride LDOUT.
  float* orow = Out + (size_t)(row0 + lhi * 8) * LDOUT + colbase + lrow;
  if (row0 + 16 <= M) {                            // full tile: branchless stores
#pragma unroll
    for (int v = 0; v < 8; ++v) orow[v * LDOUT] = c[v];
  } else {                                         // ragged tail (unused for M%16==0)
#pragma unroll
    for (int v = 0; v < 8; ++v)
      if (row0 + lhi * 8 + v < M) orow[v * LDOUT] = c[v];
  }
}

// ---- aggregation -----------------------------------------------------------
// Out[i][f] = bias[f] + dinv[i]^2 * H[i][f]   (self-loop term), pad cols -> 0
__global__ void k_agg_init(const float* __restrict__ H, float* __restrict__ Out,
                           const float* __restrict__ bias,
                           const float* __restrict__ dinv,
                           int n, int ld, int F) {
  int i = blockIdx.x * blockDim.x + threadIdx.x;
  if (i >= n * ld) return;
  int node = i / ld, f = i % ld;
  float dv = dinv[node];
  Out[i] = (f < F) ? bias[f] + dv * dv * H[i] : 0.0f;
}

// one wave per edge; lane strides features by 32 (coalesced gather + atomics)
__global__ void k_agg_edge(const float* __restrict__ H, float* __restrict__ Out,
                           const float* __restrict__ dinv,
                           const int* __restrict__ src, const int* __restrict__ dst,
                           int e, int ld, int F) {
  int eid = (blockIdx.x * blockDim.x + threadIdx.x) >> 5;
  if (eid >= e) return;
  int lane = threadIdx.x & 31;
  int s = src[eid], d = dst[eid];
  float norm = dinv[s] * dinv[d];
  for (int f = lane; f < F; f += 32)
    atomAddF(&Out[(size_t)d * ld + f], norm * H[(size_t)s * ld + f]);
}

__global__ void k_relu(float* a, int n) {
  int i = blockIdx.x * blockDim.x + threadIdx.x;
  if (i < n) a[i] = fmaxf(a[i], 0.0f);
}

// ---- log_softmax over 41 classes, one wave32 per node ----------------------
__global__ void k_logsm(const float* __restrict__ A2, float* __restrict__ out, int n) {
  int node = (blockIdx.x * blockDim.x + threadIdx.x) >> 5;
  if (node >= n) return;
  const int F = 41, LD = 48;
  int lane = threadIdx.x & 31;
  float v0 = (lane < F)      ? A2[(size_t)node * LD + lane]      : -INFINITY;
  float v1 = (lane + 32 < F) ? A2[(size_t)node * LD + lane + 32] : -INFINITY;
  float m = fmaxf(v0, v1);
#pragma unroll
  for (int off = 16; off; off >>= 1) m = fmaxf(m, __shfl_xor(m, off, 32));
  float s = ((lane < F) ? __expf(v0 - m) : 0.0f) +
            ((lane + 32 < F) ? __expf(v1 - m) : 0.0f);
#pragma unroll
  for (int off = 16; off; off >>= 1) s += __shfl_xor(s, off, 32);
  float lse = m + __logf(s);
  if (lane < F)      out[(size_t)node * F + lane]      = v0 - lse;
  if (lane + 32 < F) out[(size_t)node * F + lane + 32] = v1 - lse;
}

// ---------------------------------------------------------------------------
extern "C" void kernel_launch(void* const* d_in, const int* in_sizes, int n_in,
                              void* d_out, int out_size, void* d_ws, size_t ws_size,
                              hipStream_t stream) {
  const float* x  = (const float*)d_in[0];
  const float* W1 = (const float*)d_in[1];
  const float* b1 = (const float*)d_in[2];
  const float* W2 = (const float*)d_in[3];
  const float* b2 = (const float*)d_in[4];
  const int*   ei = (const int*)d_in[5];      // [2][E] (JAX x64 off -> int32)

  const int N = in_sizes[0] / 128;            // 100000
  const int E = in_sizes[5] / 2;              // 600000
  const int* src = ei;
  const int* dst = ei + E;

  // workspace layout (bytes); peak ~104 MB
  char* ws = (char*)d_ws;
  float*  dinv = (float*)(ws + 0);            // N f32
  __bf16* W1T  = (__bf16*)(ws + 401408);      // 128x128 bf16 (transposed)
  __bf16* W2T  = (__bf16*)(ws + 434176);      // 48x128 bf16 (transposed, padded)
  float*  H1   = (float*)(ws + 524288);       // N x 128
  float*  A1   = (float*)(ws + 52428800);     // N x 128
  float*  H2   = (float*)(ws + 524288);       // N x 48 (reuses H1 region)
  float*  A2   = (float*)(ws + 20971520);     // N x 48

  const int B = 256;

  // 1) degrees (self loop = 1) -> dinv
  k_fill1   <<<(N + B - 1) / B, B, 0, stream>>>(dinv, N);
  k_deg_edge<<<(E + B - 1) / B, B, 0, stream>>>(dinv, dst, E);
  k_rsqrt   <<<(N + B - 1) / B, B, 0, stream>>>(dinv, N);

  // 2) weights -> bf16 transposed
  k_wt<<<(128 * 128 + B - 1) / B, B, 0, stream>>>(W1, W1T, 128, 128, 128);
  k_wt<<<(48  * 128 + B - 1) / B, B, 0, stream>>>(W2, W2T, 128, 41, 48);

  // 3) layer 1: H1 = X @ W1  (8 waves/block -> 128 cols)
  gemm_wmma<128><<<(N + 15) / 16, 256, 0, stream>>>(x, W1T, H1, N);
  k_agg_init<<<(N * 128 + B - 1) / B, B, 0, stream>>>(H1, A1, b1, dinv, N, 128, 128);
  k_agg_edge<<<(E + 7) / 8, 256, 0, stream>>>(H1, A1, dinv, src, dst, E, 128, 128);
  k_relu    <<<(N * 128 + B - 1) / B, B, 0, stream>>>(A1, N * 128);

  // 4) layer 2: H2 = relu(A1) @ W2  (3 waves/block -> 48 padded cols)
  gemm_wmma<48><<<(N + 15) / 16, 96, 0, stream>>>(A1, W2T, H2, N);
  k_agg_init<<<(N * 48 + B - 1) / B, B, 0, stream>>>(H2, A2, b2, dinv, N, 48, 41);
  k_agg_edge<<<(E + 7) / 8, 256, 0, stream>>>(H2, A2, dinv, src, dst, E, 48, 41);

  // 5) log_softmax -> d_out [N][41]
  k_logsm<<<(N + 7) / 8, 256, 0, stream>>>(A2, (float*)d_out, N);
}
